// GraphConv_56779467653218
// MI455X (gfx1250) — matrix operand order repaired and checked by reading, not compiled, not run on hardware
//
#include <hip/hip_runtime.h>

// Problem constants (reference: b=16, n=2048, d=256, f=256)
#define NN 2048
#define DD 256
#define FF 256
#define NBATCH 16

// Tiling
#define BM 128
#define BN 128
#define KT 32
#define KP 40   // LDS row pitch in bf16 halves (80B rows: 16B aligned, bank spread)
#define NT 256  // threads per block = 8 waves

typedef __attribute__((ext_vector_type(16))) __bf16 v16bf;
typedef __attribute__((ext_vector_type(8)))  float  v8f;
typedef __attribute__((ext_vector_type(4)))  unsigned int u32x4;
typedef __attribute__((ext_vector_type(2)))  unsigned int u32x2;

union FragBF {
  u32x4 u[2];
  v16bf v;
};

// Round f32 -> bf16 (round-half-up; result in bits [31:16]).
__device__ __forceinline__ unsigned rnd(float f) {
  return __builtin_bit_cast(unsigned, f) + 0x8000u;
}
// Pack {hi[31:16], lo[31:16]} into one dword with a single v_perm_b32.
__device__ __forceinline__ unsigned pack2(unsigned hi, unsigned lo) {
  return __builtin_amdgcn_perm(hi, lo, 0x07060302u);
}

// ---- staging helpers (all force-inlined so LDS addrspace inference holds) ----
__device__ __forceinline__ void load_tileA(const float* __restrict__ Ab, int lda,
                                           int m0, int kk, int tid, float4 a[4]) {
#pragma unroll
  for (int c = 0; c < 4; ++c) {
    const int chunk = tid + c * NT;      // 1024 float4 chunks: BM x KT
    const int row   = chunk >> 3;        // 8 chunks per 32-float row
    const int k4    = chunk & 7;
    a[c] = *(const float4*)(Ab + (size_t)(m0 + row) * lda + kk + k4 * 4);
  }
}

__device__ __forceinline__ void load_tileB(const float* __restrict__ Bsrc, int ldb,
                                           int n0, int kk, int tid, float b[4][4]) {
#pragma unroll
  for (int c = 0; c < 4; ++c) {
    const int nloc = tid & 127;          // consecutive lanes -> consecutive n (coalesced)
    const int kg   = (tid >> 7) + 2 * c; // 8 groups of 4 k-rows
    const float* p = Bsrc + (size_t)(kk + kg * 4) * ldb + n0 + nloc;
#pragma unroll
    for (int j = 0; j < 4; ++j) b[c][j] = p[(size_t)j * ldb];
  }
}

__device__ __forceinline__ void store_tiles(unsigned short* As, unsigned short* Bs,
                                            int tid, const float4 a[4], const float b[4][4]) {
#pragma unroll
  for (int c = 0; c < 4; ++c) {          // A: row-major bf16 [BM][KP]
    const int chunk = tid + c * NT;
    const int row   = chunk >> 3;
    const int k4    = chunk & 7;
    u32x2 p;
    p.x = pack2(rnd(a[c].y), rnd(a[c].x));
    p.y = pack2(rnd(a[c].w), rnd(a[c].z));
    *(u32x2*)(&As[row * KP + k4 * 4]) = p;
  }
#pragma unroll
  for (int c = 0; c < 4; ++c) {          // B: transposed bf16 [BN][KP] (n-major)
    const int nloc = tid & 127;
    const int kg   = (tid >> 7) + 2 * c;
    u32x2 p;
    p.x = pack2(rnd(b[c][1]), rnd(b[c][0]));
    p.y = pack2(rnd(b[c][3]), rnd(b[c][2]));
    *(u32x2*)(&Bs[nloc * KP + kg * 4]) = p;
  }
}

// Fragment loads (ISA 16-bit A 16x32 / B 32x16 layouts) + 8 WMMAs.
__device__ __forceinline__ void mma_step(const unsigned short* As, const unsigned short* Bs,
                                         int waveM, int waveN, int lrow, int hi,
                                         v8f acc[2][4]) {
  FragBF a[2], b[4];
#pragma unroll
  for (int i = 0; i < 2; ++i) {
    const unsigned short* p = &As[(waveM * 32 + i * 16 + lrow) * KP];
    a[i].u[0] = *(const u32x4*)(p + 8 * hi);       // K = 8*hi .. +7
    a[i].u[1] = *(const u32x4*)(p + 16 + 8 * hi);  // K = 16+8*hi .. +7
  }
#pragma unroll
  for (int j = 0; j < 4; ++j) {
    const unsigned short* p = &Bs[(waveN * 64 + j * 16 + lrow) * KP + 8 * hi];
    b[j].u[0] = *(const u32x4*)(p);                // K = 8*hi .. +7
    b[j].u[1] = *(const u32x4*)(p + 16);           // K = 16+8*hi .. +7
  }
#pragma unroll
  for (int i = 0; i < 2; ++i)
#pragma unroll
    for (int j = 0; j < 4; ++j)
      acc[i][j] = __builtin_amdgcn_wmma_f32_16x16x32_bf16(
          false, a[i].v, false, b[j].v, (short)0, acc[i][j], false, false);
}

// ---------------------------------------------------------------------------
// Kernel 1: agg[b] = A[b] (n x n) @ F[b] (n x d), fp32 out to workspace.
// ---------------------------------------------------------------------------
__global__ __launch_bounds__(NT, 2)
void k_agg(const float* __restrict__ Aadj, const float* __restrict__ Feat,
           float* __restrict__ Agg) {
  __shared__ unsigned short As[2][BM * KP];
  __shared__ unsigned short Bs[2][BN * KP];

  const int tid   = threadIdx.x;
  const int batch = blockIdx.z;
  const int m0    = blockIdx.x * BM;
  const int n0    = blockIdx.y * BN;

  const float* Ab = Aadj + (size_t)batch * NN * NN;
  const float* Fb = Feat + (size_t)batch * NN * DD;
  float*       Ob = Agg  + (size_t)batch * NN * DD;

  const int wave  = tid >> 5;
  const int lane  = tid & 31;
  const int hi    = lane >> 4;
  const int lrow  = lane & 15;
  const int waveM = wave & 3;
  const int waveN = wave >> 2;

  v8f acc[2][4] = {};
  float4 aR[4];
  float  bR[4][4];

  // prologue: stage tile 0 into buffer 0
  load_tileA(Ab, NN, m0, 0, tid, aR);
  load_tileB(Fb, DD, n0, 0, tid, bR);
  store_tiles(As[0], Bs[0], tid, aR, bR);
  __syncthreads();

  const int NITER = NN / KT;
  int cur = 0;
  for (int it = 0; it < NITER; ++it) {
    const bool nxt = (it + 1) < NITER;
    if (nxt) {                                   // global loads in flight over WMMA
      load_tileA(Ab, NN, m0, (it + 1) * KT, tid, aR);
      load_tileB(Fb, DD, n0, (it + 1) * KT, tid, bR);
    }
    if (it + 2 < NITER)                          // global_prefetch_b8 on the HBM A-stream
      __builtin_prefetch(Ab + (size_t)(m0 + (tid >> 1)) * NN + (it + 2) * KT + (tid & 1) * 16, 0, 1);

    mma_step(As[cur], Bs[cur], waveM, waveN, lrow, hi, acc);

    if (nxt) store_tiles(As[cur ^ 1], Bs[cur ^ 1], tid, aR, bR);
    __syncthreads();                             // single barrier: r/w buffers disjoint
    cur ^= 1;
  }

  // epilogue: C layout VGPR r -> M = r + 8*hi, N = lrow
#pragma unroll
  for (int i = 0; i < 2; ++i)
#pragma unroll
    for (int j = 0; j < 4; ++j) {
      const int col = n0 + waveN * 64 + j * 16 + lrow;
#pragma unroll
      for (int r = 0; r < 8; ++r) {
        const int row = m0 + waveM * 32 + i * 16 + r + 8 * hi;
        Ob[(size_t)row * DD + col] = acc[i][j][r];
      }
    }
}

// ---------------------------------------------------------------------------
// Kernel 2: out = relu([F | agg] @ [W1; W2] + bias)  (single K=2d GEMM)
// ---------------------------------------------------------------------------
__global__ __launch_bounds__(NT, 2)
void k_out(const float* __restrict__ Feat, const float* __restrict__ Agg,
           const float* __restrict__ W, const float* __restrict__ Bias,
           float* __restrict__ Out) {
  __shared__ unsigned short As[2][BM * KP];
  __shared__ unsigned short Bs[2][BN * KP];

  const int tid   = threadIdx.x;
  const int batch = blockIdx.z;
  const int m0    = blockIdx.x * BM;
  const int n0    = blockIdx.y * BN;

  const float* Fb = Feat + (size_t)batch * NN * DD;
  const float* Gb = Agg  + (size_t)batch * NN * DD;
  float*       Ob = Out  + (size_t)batch * NN * FF;

  const int wave  = tid >> 5;
  const int lane  = tid & 31;
  const int hi    = lane >> 4;
  const int lrow  = lane & 15;
  const int waveM = wave & 3;
  const int waveN = wave >> 2;

  v8f acc[2][4] = {};
  float4 aR[4];
  float  bR[4][4];

  // prologue: tile 0 comes from Feat / W rows 0..KT
  load_tileA(Fb, DD, m0, 0, tid, aR);
  load_tileB(W, FF, n0, 0, tid, bR);
  store_tiles(As[0], Bs[0], tid, aR, bR);
  __syncthreads();

  const int NITER = (2 * DD) / KT;
  int cur = 0;
  for (int it = 0; it < NITER; ++it) {
    const bool nxt = (it + 1) < NITER;
    if (nxt) {
      const int kn = (it + 1) * KT;
      const float* srcA = (kn < DD) ? Fb : Gb;   // K concat: [F | agg]
      const int    ks   = (kn < DD) ? kn : kn - DD;
      load_tileA(srcA, DD, m0, ks, tid, aR);
      load_tileB(W, FF, n0, kn, tid, bR);        // W rows = [W1; W2]
    }

    mma_step(As[cur], Bs[cur], waveM, waveN, lrow, hi, acc);

    if (nxt) store_tiles(As[cur ^ 1], Bs[cur ^ 1], tid, aR, bR);
    __syncthreads();
    cur ^= 1;
  }

#pragma unroll
  for (int i = 0; i < 2; ++i)
#pragma unroll
    for (int j = 0; j < 4; ++j) {
      const int   col = n0 + waveN * 64 + j * 16 + lrow;
      const float bv  = Bias[col];
#pragma unroll
      for (int r = 0; r < 8; ++r) {
        const int row = m0 + waveM * 32 + i * 16 + r + 8 * hi;
        Ob[(size_t)row * FF + col] = fmaxf(acc[i][j][r] + bv, 0.0f);
      }
    }
}

// ---------------------------------------------------------------------------
// Host launcher. Inputs: features[b,n,d] f32, A[b,n,n] f32, weight[2d,f] f32,
// bias[f] f32. Output: [b,n,f] f32. Workspace: agg[b,n,d] f32 (32 MB).
// ---------------------------------------------------------------------------
extern "C" void kernel_launch(void* const* d_in, const int* in_sizes, int n_in,
                              void* d_out, int out_size, void* d_ws, size_t ws_size,
                              hipStream_t stream) {
  const float* feat = (const float*)d_in[0];
  const float* Aadj = (const float*)d_in[1];
  const float* W    = (const float*)d_in[2];
  const float* bias = (const float*)d_in[3];
  float* out = (float*)d_out;
  float* agg = (float*)d_ws;

  dim3 block(NT, 1, 1);
  dim3 grid1(NN / BM, DD / BN, NBATCH);   // 16 x 2 x 16
  k_agg<<<grid1, block, 0, stream>>>(Aadj, feat, agg);

  dim3 grid2(NN / BM, FF / BN, NBATCH);   // 16 x 2 x 16
  k_out<<<grid2, block, 0, stream>>>(feat, agg, W, bias, out);
}
